// SphericalSiren_15745350107215
// MI455X (gfx1250) — matrix operand order repaired and checked by reading, not compiled, not run on hardware
//
#include <hip/hip_runtime.h>
#include <math.h>

typedef _Float16 v16h __attribute__((ext_vector_type(16)));
typedef _Float16 v8h  __attribute__((ext_vector_type(8)));
typedef float    v8f  __attribute__((ext_vector_type(8)));

#define LMAX 10
#define N_SH 121          // (LMAX+1)^2
#define KPAD 128          // padded K
#define NEURONS 128
#define PTS_PER_BLK 64    // 2 waves x 2 row-tiles x 16 rows
#define THREADS 64
#define WSTRIDE (NEURONS * KPAD)   // 16384 halves per layer

// ---------------------------------------------------------------------------
// Kernel A1: layer-0 weights: coeffs[k][c] = sum_j sh_W[k][c][j] + sh_b[k][c],
// packed f16, K padded 121 -> 128 with zeros.  grid = 128 x 128.
// ---------------------------------------------------------------------------
__global__ void sh_coeffs_kernel(const float* __restrict__ shW,
                                 const float* __restrict__ shb,
                                 _Float16* __restrict__ wAll) {
    int n = blockIdx.x;      // neuron 0..127
    int c = threadIdx.x;     // sh column 0..127
    if (c < N_SH) {
        float s = shb[n * N_SH + c];
        const float* row = shW + ((size_t)n * N_SH + c) * N_SH;
        #pragma unroll 11
        for (int j = 0; j < N_SH; ++j) s += row[j];
        wAll[n * KPAD + c] = (_Float16)s;
    } else {
        wAll[n * KPAD + c] = (_Float16)0.f;
    }
}

// ---------------------------------------------------------------------------
// Kernel A2: hidden-layer weights f32 -> f16, one pass.  3*16384 elements,
// 4 per thread.  grid = 48 x 256.
// ---------------------------------------------------------------------------
__global__ void hid_pack_kernel(const float* __restrict__ hidW,
                                _Float16* __restrict__ wAll) {
    int i = (blockIdx.x * 256 + threadIdx.x) * 4;
    const float4 v = *(const float4*)(hidW + i);
    _Float16* dst = wAll + WSTRIDE + i;          // layers 1..3 after coeffs
    dst[0] = (_Float16)v.x;
    dst[1] = (_Float16)v.y;
    dst[2] = (_Float16)v.z;
    dst[3] = (_Float16)v.w;
}

// SH normalization K(l,m) = sqrt((2l+1)/(4pi) * (l-m)!/(l+m)!)
__device__ inline float shK(int l, int m) {
    float r = 1.0f;
    for (int i = l - m + 1; i <= l + m; ++i) r *= (float)i;
    return sqrtf((float)(2 * l + 1) / (4.0f * (float)M_PI) / r);
}

// ---------------------------------------------------------------------------
// Kernel B: per-block MLP over 64 points.  2 waves, each owning two 16-row
// tiles so every B fragment feeds two v_wmma_f32_16x16x32_f16.
// ---------------------------------------------------------------------------
__global__ void __launch_bounds__(THREADS)
siren_kernel(const float* __restrict__ coords,
             const _Float16* __restrict__ wAll,  // f16 weights [4][128][128]
             const float* __restrict__ finW,     // [128]
             const float* __restrict__ finb,     // [1]
             float* __restrict__ out) {
    __shared__ _Float16 Wlds[WSTRIDE];             // 32 KB: layer weights W[n][k]
    __shared__ _Float16 Xlds[PTS_PER_BLK * KPAD];  // 16 KB: activations / Y basis
    __shared__ float    Flds[NEURONS];             // final weights

    const int tid  = threadIdx.x;
    const int lane = tid & 31;
    const int wv   = tid >> 5;        // wave 0..1, owns rows [wv*32, wv*32+32)
    const int hl   = lane >> 4;       // lane half (K-range selector)
    const int nl   = lane & 15;       // row/col within tile

    Flds[tid] = finW[tid];            // 64 threads, two each
    Flds[tid + 64] = finW[tid + 64];

    // ---- Phase 1: spherical-harmonics basis, one point per thread (f16) ----
    {
        const int gp = blockIdx.x * PTS_PER_BLK + tid;
        const float theta = coords[2 * gp];
        const float phi   = coords[2 * gp + 1];
        const float ct = cosf(theta), st = sinf(theta);
        _Float16* Yr = &Xlds[tid * KPAD];
        #pragma unroll
        for (int c = N_SH; c < KPAD; ++c) Yr[c] = (_Float16)0.f;
        float pmm = 1.0f;                     // P[m][m]
        for (int m = 0; m <= LMAX; ++m) {
            const float cm = cosf((float)m * phi);
            const float sm = sinf((float)m * phi);
            float p0 = pmm, p1 = 0.f;         // P[l-2][m], P[l-1][m]
            for (int l = m; l <= LMAX; ++l) {
                float p;
                if (l == m)            p = p0;
                else if (l == m + 1) { p1 = (float)(2 * m + 1) * ct * p0; p = p1; }
                else {
                    p = ((float)(2 * l - 1) * ct * p1 - (float)(l + m - 1) * p0)
                        / (float)(l - m);
                    p0 = p1; p1 = p;
                }
                const float K = shK(l, m);
                const int base = l * l + l;
                if (m == 0) {
                    Yr[base] = (_Float16)(K * p);
                } else {
                    Yr[base + m] = (_Float16)(1.41421356237f * K * cm * p);
                    Yr[base - m] = (_Float16)(1.41421356237f * K * sm * p);
                }
            }
            pmm *= (float)(2 * m + 1) * st;   // -> P[m+1][m+1]
        }
    }

    // ---- Phase 2: four WMMA layers (121->128, then 3 x 128->128), g(x)=1-x^2/2
    for (int layer = 0; layer < 4; ++layer) {
        __syncthreads();   // prior Xlds writes visible; prior Wlds reads done
        {   // pure f16 copy: global_load_b128 -> ds_store_b128
            const uint4* src = (const uint4*)(wAll + (size_t)layer * WSTRIDE);
            uint4* dst = (uint4*)Wlds;
            #pragma unroll 4
            for (int i = tid; i < WSTRIDE / 8; i += THREADS) dst[i] = src[i];
        }
        if (layer < 3)     // pull next layer's weights toward L2/WGP early
            __builtin_prefetch(wAll + (size_t)(layer + 1) * WSTRIDE, 0, 0);
        __syncthreads();

        // A fragments: two 16x32 row tiles per wave.
        // lane L: M = L&15; VGPR v -> K = 16*hl + 2v (16 contiguous halves)
        v16h a[2][4];
        #pragma unroll
        for (int i = 0; i < 2; ++i) {
            #pragma unroll
            for (int kb = 0; kb < 4; ++kb) {
                const _Float16* row = &Xlds[(wv * 32 + i * 16 + nl) * KPAD + kb * 32];
                v8h c0 = *(const v8h*)(row + hl * 16);
                v8h c1 = *(const v8h*)(row + hl * 16 + 8);
                a[i][kb] = __builtin_shufflevector(c0, c1, 0, 1, 2, 3, 4, 5, 6, 7,
                                                           8, 9, 10, 11, 12, 13, 14, 15);
            }
        }

        // B fragment 32x16: B[k][n] = W[n][k]; lane L: N = L&15,
        // K = kb*32 + hl*16 + (2v, 2v+1) -> 16 contiguous halves.
        // Double-buffered so tile t+1's loads are in flight during tile t's WMMAs.
        v8f  acc[2][8] = {};
        v16h bb[2][4];
        #pragma unroll
        for (int kb = 0; kb < 4; ++kb)
            bb[0][kb] = *(const v16h*)&Wlds[nl * KPAD + kb * 32 + hl * 16];

        #pragma unroll
        for (int t = 0; t < 8; ++t) {              // 8 column tiles of 16 neurons
            const int cur = t & 1, nxt = cur ^ 1;
            if (t < 7) {
                const int n0 = (t + 1) * 16;
                #pragma unroll
                for (int kb = 0; kb < 4; ++kb)
                    bb[nxt][kb] = *(const v16h*)&Wlds[(n0 + nl) * KPAD + kb * 32 + hl * 16];
            }
            #pragma unroll
            for (int kb = 0; kb < 4; ++kb) {
                acc[0][t] = __builtin_amdgcn_wmma_f32_16x16x32_f16(
                    false, a[0][kb], false, bb[cur][kb], (short)0, acc[0][t], false, false);
                acc[1][t] = __builtin_amdgcn_wmma_f32_16x16x32_f16(
                    false, a[1][kb], false, bb[cur][kb], (short)0, acc[1][t], false, false);
            }
        }

        __syncthreads();   // all Xlds/Wlds reads complete before overwrite
        // g(x) = 1 - x^2/2, write back f16 per C/D layout:
        // VGPR r: lanes 0-15 -> M=r, lanes 16-31 -> M=8+r; N = lane&15
        #pragma unroll
        for (int i = 0; i < 2; ++i) {
            #pragma unroll
            for (int t = 0; t < 8; ++t) {
                #pragma unroll
                for (int r = 0; r < 8; ++r) {
                    float x = acc[i][t][r];
                    x = 1.0f - 0.5f * x * x;
                    Xlds[(wv * 32 + i * 16 + hl * 8 + r) * KPAD + t * 16 + nl] = (_Float16)x;
                }
            }
        }
    }
    __syncthreads();

    // ---- Phase 3: final dot (128 -> 1) + bias, one point per thread ----
    {
        const _Float16* xr = &Xlds[tid * KPAD];
        float s = finb[0];
        #pragma unroll 8
        for (int n = 0; n < NEURONS; ++n) s += (float)xr[n] * Flds[n];
        out[blockIdx.x * PTS_PER_BLK + tid] = s;
    }
}

extern "C" void kernel_launch(void* const* d_in, const int* in_sizes, int n_in,
                              void* d_out, int out_size, void* d_ws, size_t ws_size,
                              hipStream_t stream) {
    const float* coords = (const float*)d_in[0];   // [N,2]
    const float* shW    = (const float*)d_in[1];   // [128,121,121]
    const float* shb    = (const float*)d_in[2];   // [128,121]
    const float* hidW   = (const float*)d_in[3];   // [3,128,128]
    const float* finW   = (const float*)d_in[4];   // [1,128]
    const float* finb   = (const float*)d_in[5];   // [1]
    float* out = (float*)d_out;

    _Float16* wAll = (_Float16*)d_ws;              // 4 * 16384 f16 = 128 KB

    sh_coeffs_kernel<<<dim3(NEURONS), dim3(KPAD), 0, stream>>>(shW, shb, wAll);
    hid_pack_kernel<<<dim3(48), dim3(256), 0, stream>>>(hidW, wAll);

    const int npts = out_size;                     // 262144
    const int nblk = (npts + PTS_PER_BLK - 1) / PTS_PER_BLK;
    siren_kernel<<<dim3(nblk), dim3(THREADS), 0, stream>>>(
        coords, wAll, finW, finb, out);
}
